// E2EStackedBiRNNMpPoolAttention_82437602279862
// MI455X (gfx1250) — compile-verified, matrix operand order checked by reading, not run on hardware
//
#include <hip/hip_runtime.h>
#include <hip/hip_bf16.h>
#include <math.h>

typedef __attribute__((ext_vector_type(16))) _Float16 v16h;
typedef __attribute__((ext_vector_type(8)))  _Float16 v8h;
typedef __attribute__((ext_vector_type(8)))  float    v8f;

#define B_    64
#define T_    100
#define EMB_  256
#define U_    256
#define P_    256
#define OUT_  4
#define BT    (B_ * T_)     // 6400
#define K0P   288           // layer-0 input dim padded 258 -> 288
#define G3    (3 * U_)      // 768
#define TPAD  112           // T padded to mult of 16
#define SPAD  128           // attention K padded to mult of 32
#define BG    16            // batch rows per scan block
#define NBG   (B_ / BG)     // 4 batch groups

// ---------------------------------------------------------------------------
// WMMA fragment loader: 16x32 f16 tile operand for v_wmma_f32_16x16x32_f16.
// Lane L (m or n = L%16, half = L/16) needs K in [8h,8h+8) U [16+8h,16+8h+8):
// two contiguous 16-byte chunks -> two b128 loads.
// ---------------------------------------------------------------------------
__device__ inline v16h load_frag(const _Float16* __restrict__ base, int ld, int k0, int lane) {
  const int r = lane & 15;
  const int h = lane >> 4;
  const _Float16* p = base + (size_t)r * ld + (k0 + 8 * h);
  v8h lo = *(const v8h*)(p);
  v8h hi = *(const v8h*)(p + 16);
  v16h f;
#pragma unroll
  for (int i = 0; i < 8; ++i) { f[i] = lo[i]; f[8 + i] = hi[i]; }
  return f;
}

__device__ inline v8f zero8() {
  v8f z;
#pragma unroll
  for (int i = 0; i < 8; ++i) z[i] = 0.f;
  return z;
}

// ---------------------------------------------------------------------------
// Generic batched WMMA GEMM: C[b] = A[b](MxK) * Bt[b](NxK)^T, optional bias +
// relu, f32 and/or f16 outputs. A row-major [M][K], Bt row-major [N][K].
// Each wave computes one 16 x (16*NW) tile.
// ---------------------------------------------------------------------------
template <int NW>
__global__ void __launch_bounds__(256) gemm_wmma(
    const _Float16* __restrict__ A, const _Float16* __restrict__ Bt,
    float* __restrict__ Cf, _Float16* __restrict__ Ch,
    int M, int N, int K, int lda, int ldb, int ldc,
    long sA, long sB, long sC, int nbatch,
    int Mvalid, int Nvalid, const float* __restrict__ bias, int relu) {
  const int lane = threadIdx.x & 31;
  const int wave = threadIdx.x >> 5;
  const int mt = M / 16;
  const int ngrp = N / (16 * NW);
  const long tilesPerBatch = (long)mt * ngrp;
  const long total = tilesPerBatch * nbatch;
  const long tile = (long)blockIdx.x * (blockDim.x >> 5) + wave;
  if (tile >= total) return;  // wave-uniform exit (EXEC stays all-ones for WMMA)

  const int b = (int)(tile / tilesPerBatch);
  const long t2 = tile % tilesPerBatch;
  const int im = (int)(t2 / ngrp);
  const int in = (int)(t2 % ngrp);

  const _Float16* Ab = A + (long)b * sA + (long)im * 16 * lda;
  const _Float16* Bb = Bt + (long)b * sB + (long)in * (16 * NW) * ldb;

  v8f acc[NW];
#pragma unroll
  for (int i = 0; i < NW; ++i) acc[i] = zero8();

  for (int k0 = 0; k0 < K; k0 += 32) {
    v16h a = load_frag(Ab, lda, k0, lane);
#pragma unroll
    for (int i = 0; i < NW; ++i) {
      v16h bf = load_frag(Bb + (long)i * 16 * ldb, ldb, k0, lane);
      acc[i] = __builtin_amdgcn_wmma_f32_16x16x32_f16(
          false, a, false, bf, (short)0, acc[i], false, false);
    }
  }

  const int r0 = im * 16 + 8 * (lane >> 4);
  const int c00 = in * 16 * NW + (lane & 15);
#pragma unroll
  for (int i = 0; i < NW; ++i) {
    const int c = c00 + i * 16;
    const float bv = bias ? bias[c] : 0.f;
#pragma unroll
    for (int v = 0; v < 8; ++v) {
      const int r = r0 + v;
      if (r < Mvalid && c < Nvalid) {
        float val = acc[i][v] + bv;
        if (relu) val = fmaxf(val, 0.f);
        const long o = (long)b * sC + (long)r * ldc + c;
        if (Cf) Cf[o] = val;
        if (Ch) Ch[o] = (_Float16)val;
      }
    }
  }
}

// ---------------------------------------------------------------------------
// Persistent GRU scan. Batch rows are independent across the whole scan, so
// grid = 2 dirs x 4 batch-groups of 16 rows = 8 blocks (8 WGPs busy during
// the latency-critical recurrence). 384 threads = 12 waves; per step each
// wave computes one 16x64 tile of gh(16x768) = h(16x256) @ Whh^T, then all
// threads apply the fused gate update. LDS/block = 8K (h f16) + 16K (h f32)
// + 48K (gh) = 72KB. Whh streams from L2 every step; an asm memory clobber
// stops the compiler from hoisting those loop-invariant fragment loads out
// of the 100-iteration time loop (which previously caused scratch spills).
// ---------------------------------------------------------------------------
__global__ void __launch_bounds__(384, 1) gru_scan(
    const float* __restrict__ xpF, const float* __restrict__ xpB,
    const _Float16* __restrict__ WhhF, const _Float16* __restrict__ WhhB,
    const float* __restrict__ bhhF, const float* __restrict__ bhhB,
    _Float16* __restrict__ H) {
  const int dir = blockIdx.x >> 2;       // 0..1
  const int bg = blockIdx.x & 3;         // 0..3 (batch rows bg*16 .. bg*16+15)
  const float* xp = dir ? xpB : xpF;
  const _Float16* Whh = dir ? WhhB : WhhF;
  const float* bhh = dir ? bhhB : bhhF;
  const int dirOff = dir * U_;

  extern __shared__ char smem[];
  _Float16* hf16 = (_Float16*)smem;                     // 16*256 f16  (8KB)
  float* hf32 = (float*)(smem + BG * U_ * 2);           // 16*256 f32 (16KB)
  float* gh = (float*)(smem + BG * U_ * 2 + BG * U_ * 4);  // 16*768 f32 (48KB)

  const int tid = threadIdx.x;
  const int lane = tid & 31;
  const int wave = tid >> 5;             // 0..11 -> n-group (64 cols each)

  for (int i = tid; i < BG * U_; i += blockDim.x) {
    hf32[i] = 0.f;
    hf16[i] = (_Float16)0.f;
  }
  __syncthreads();

  for (int s = 0; s < T_; ++s) {
    asm volatile("" ::: "memory");  // keep Whh fragment loads inside this loop
    const int t = dir ? (T_ - 1 - s) : s;

    // gh = h @ Whh^T : 12 n-groups of 64, one per wave, M=16, K=256
    {
      v8f acc[4];
#pragma unroll
      for (int i = 0; i < 4; ++i) acc[i] = zero8();
      const _Float16* Bbase = Whh + (size_t)(wave * 64) * U_;
      for (int k0 = 0; k0 < U_; k0 += 32) {
        v16h a = load_frag(hf16, U_, k0, lane);
#pragma unroll
        for (int i = 0; i < 4; ++i) {
          v16h bf = load_frag(Bbase + (size_t)(i * 16) * U_, U_, k0, lane);
          acc[i] = __builtin_amdgcn_wmma_f32_16x16x32_f16(
              false, a, false, bf, (short)0, acc[i], false, false);
        }
      }
      const int r0 = 8 * (lane >> 4);
      const int c0 = wave * 64 + (lane & 15);
#pragma unroll
      for (int i = 0; i < 4; ++i)
#pragma unroll
        for (int v = 0; v < 8; ++v)
          gh[(r0 + v) * G3 + c0 + i * 16] = acc[i][v];
    }
    __syncthreads();

    // fused GRU gates over 16 rows x 256 units
    for (int i = tid; i < BG * U_; i += blockDim.x) {
      const int lb = i >> 8;             // local row 0..15
      const int b = bg * BG + lb;        // global batch row
      const int u = i & 255;
      const float* xrow = xp + ((size_t)(b * T_ + t)) * G3;
      const float ir = xrow[u];
      const float iz = xrow[U_ + u];
      const float inn = xrow[2 * U_ + u];
      const float hr = gh[lb * G3 + u] + bhh[u];
      const float hz = gh[lb * G3 + U_ + u] + bhh[U_ + u];
      const float hn = gh[lb * G3 + 2 * U_ + u] + bhh[2 * U_ + u];
      const float r = 1.f / (1.f + expf(-(ir + hr)));
      const float z = 1.f / (1.f + expf(-(iz + hz)));
      const float n = tanhf(inn + r * hn);
      const float hnew = (1.f - z) * n + z * hf32[i];
      hf32[i] = hnew;
      hf16[i] = (_Float16)hnew;
      H[((size_t)(b * T_ + t)) * (2 * U_) + dirOff + u] = (_Float16)hnew;
    }
    __syncthreads();
  }
}

// ---------------------------------------------------------------------------
// Small elementwise / prep kernels
// ---------------------------------------------------------------------------
__global__ void f32_to_f16_pad(const float* __restrict__ src, _Float16* __restrict__ dst,
                               int rows, int srcK, int dstK) {
  const int idx = blockIdx.x * blockDim.x + threadIdx.x;
  if (idx >= rows * dstK) return;
  const int r = idx / dstK, c = idx % dstK;
  dst[idx] = (c < srcK) ? (_Float16)src[(size_t)r * srcK + c] : (_Float16)0.f;
}

__global__ void watt_transpose(const float* __restrict__ W, _Float16* __restrict__ Wt) {
  const int idx = blockIdx.x * blockDim.x + threadIdx.x;
  if (idx >= P_ * P_) return;
  const int q = idx / P_, p = idx % P_;
  Wt[idx] = (_Float16)W[p * P_ + q];  // Wt[q][p] = W[p][q]
}

__global__ void embed_concat(const int* __restrict__ words, const float* __restrict__ is_pred,
                             const float* __restrict__ is_target, const float* __restrict__ emb,
                             _Float16* __restrict__ X0) {
  const int idx = blockIdx.x * blockDim.x + threadIdx.x;
  if (idx >= BT * K0P) return;
  const int r = idx / K0P, c = idx % K0P;
  float v;
  if (c < EMB_)            v = emb[(size_t)words[r] * EMB_ + c];
  else if (c == EMB_)      v = is_pred[r];
  else if (c == EMB_ + 1)  v = is_target[r];
  else                     v = 0.f;
  X0[idx] = (_Float16)v;
}

__global__ void cmax_kernel(const float* __restrict__ C, float* __restrict__ Cmax) {
  const int idx = blockIdx.x * blockDim.x + threadIdx.x;
  if (idx >= T_ * P_) return;
  const int t = idx / P_, p = idx % P_;
  float m = -1e30f;
  for (int b = 0; b < B_; ++b) m = fmaxf(m, C[((size_t)(b * T_ + t)) * P_ + p]);
  Cmax[idx] = m;
}

__global__ void pooled_kernel(const float* __restrict__ A, const float* __restrict__ Cmax,
                              const float* __restrict__ b_lin,
                              float* __restrict__ pf, _Float16* __restrict__ ph) {
  const int idx = blockIdx.x * blockDim.x + threadIdx.x;
  if (idx >= BT * P_) return;
  const int row = idx / P_, p = idx % P_, t = row % T_;
  float v = A[idx] + Cmax[t * P_ + p] + b_lin[p];
  v = fmaxf(v, 0.f);
  pf[idx] = v;
  ph[idx] = (_Float16)v;
}

__global__ void softmax_att(const float* __restrict__ scores, _Float16* __restrict__ att) {
  __shared__ float red[128];
  const int row = blockIdx.x;  // b*T + t
  const int b = row / T_, t = row % T_;
  const int s = threadIdx.x;   // 0..127
  const float v = (s < T_) ? scores[(size_t)row * T_ + s] : -1e30f;
  red[s] = v;
  __syncthreads();
  for (int o = 64; o > 0; o >>= 1) {
    if (s < o) red[s] = fmaxf(red[s], red[s + o]);
    __syncthreads();
  }
  const float m = red[0];
  __syncthreads();
  const float e = (s < T_) ? expf(v - m) : 0.f;
  red[s] = e;
  __syncthreads();
  for (int o = 64; o > 0; o >>= 1) {
    if (s < o) red[s] += red[s + o];
    __syncthreads();
  }
  const float inv = 1.f / red[0];
  att[((size_t)(b * TPAD + t)) * SPAD + s] = (_Float16)(e * inv);
}

__global__ void pooledT_kernel(const float* __restrict__ pf, _Float16* __restrict__ pT) {
  const int idx = blockIdx.x * blockDim.x + threadIdx.x;
  if (idx >= B_ * P_ * SPAD) return;
  const int b = idx / (P_ * SPAD);
  const int rem = idx % (P_ * SPAD);
  const int p = rem / SPAD, s = rem % SPAD;
  const float v = (s < T_) ? pf[((size_t)(b * T_ + s)) * P_ + p] : 0.f;
  pT[idx] = (_Float16)v;
}

__global__ void comb_kernel(const float* __restrict__ pf, const float* __restrict__ attend,
                            _Float16* __restrict__ comb) {
  const int idx = blockIdx.x * blockDim.x + threadIdx.x;
  if (idx >= BT * 2 * P_) return;
  const int row = idx / (2 * P_), c = idx % (2 * P_);
  const float v = (c < P_) ? pf[(size_t)row * P_ + c] : attend[(size_t)row * P_ + (c - P_)];
  comb[idx] = (_Float16)v;
}

__global__ void final_logits(const _Float16* __restrict__ hid, const float* __restrict__ Wout,
                             const float* __restrict__ bout, float* __restrict__ out) {
  const int row = blockIdx.x * blockDim.x + threadIdx.x;
  if (row >= BT) return;
  const _Float16* hrow = hid + (size_t)row * P_;
  float l[OUT_];
#pragma unroll
  for (int o = 0; o < OUT_; ++o) {
    float acc = bout[o];
    const float* w = Wout + o * P_;
    for (int k = 0; k < P_; ++k) acc += (float)hrow[k] * w[k];
    l[o] = acc;
  }
  float m = l[0];
#pragma unroll
  for (int o = 1; o < OUT_; ++o) m = fmaxf(m, l[o]);
  float ssum = 0.f;
#pragma unroll
  for (int o = 0; o < OUT_; ++o) ssum += expf(l[o] - m);
  const float lse = logf(ssum);
#pragma unroll
  for (int o = 0; o < OUT_; ++o) out[row * OUT_ + o] = l[o] - m - lse;
}

// ---------------------------------------------------------------------------
// Host driver
// ---------------------------------------------------------------------------
extern "C" void kernel_launch(void* const* d_in, const int* in_sizes, int n_in,
                              void* d_out, int out_size, void* d_ws, size_t ws_size,
                              hipStream_t stream) {
  (void)in_sizes; (void)n_in; (void)out_size; (void)ws_size;
  const int*   words     = (const int*)d_in[0];
  const float* is_pred   = (const float*)d_in[1];
  const float* is_target = (const float*)d_in[2];
  const float* emb       = (const float*)d_in[3];
  const float* Wf0 = (const float*)d_in[4],  *Uf0 = (const float*)d_in[5];
  const float* bif0= (const float*)d_in[6],  *bhf0= (const float*)d_in[7];
  const float* Wb0 = (const float*)d_in[8],  *Ub0 = (const float*)d_in[9];
  const float* bib0= (const float*)d_in[10], *bhb0= (const float*)d_in[11];
  const float* Wf1 = (const float*)d_in[12], *Uf1 = (const float*)d_in[13];
  const float* bif1= (const float*)d_in[14], *bhf1= (const float*)d_in[15];
  const float* Wb1 = (const float*)d_in[16], *Ub1 = (const float*)d_in[17];
  const float* bib1= (const float*)d_in[18], *bhb1= (const float*)d_in[19];
  const float* W_lin = (const float*)d_in[20], *b_lin  = (const float*)d_in[21];
  const float* W_att = (const float*)d_in[22];
  const float* W_lin2= (const float*)d_in[23], *b_lin2 = (const float*)d_in[24];
  const float* W_out = (const float*)d_in[25], *b_out  = (const float*)d_in[26];

  char* wsp = (char*)d_ws;
  auto alloc = [&](size_t bytes) -> char* {
    char* p = wsp;
    wsp += (bytes + 255) & ~(size_t)255;
    return p;
  };
  _Float16* X0     = (_Float16*)alloc((size_t)BT * K0P * 2);
  float*    xpFb   = (float*)   alloc((size_t)BT * G3 * 4);
  float*    xpBb   = (float*)   alloc((size_t)BT * G3 * 4);
  _Float16* H1     = (_Float16*)alloc((size_t)BT * 2 * U_ * 2);
  _Float16* H2     = (_Float16*)alloc((size_t)BT * 2 * U_ * 2);
  _Float16* Wih0F  = (_Float16*)alloc((size_t)G3 * K0P * 2);
  _Float16* Wih0B  = (_Float16*)alloc((size_t)G3 * K0P * 2);
  _Float16* Whh0F  = (_Float16*)alloc((size_t)G3 * U_ * 2);
  _Float16* Whh0B  = (_Float16*)alloc((size_t)G3 * U_ * 2);
  _Float16* Wih1F  = (_Float16*)alloc((size_t)G3 * 512 * 2);
  _Float16* Wih1B  = (_Float16*)alloc((size_t)G3 * 512 * 2);
  _Float16* Whh1F  = (_Float16*)alloc((size_t)G3 * U_ * 2);
  _Float16* Whh1B  = (_Float16*)alloc((size_t)G3 * U_ * 2);
  _Float16* WlinH  = (_Float16*)alloc((size_t)P_ * 1024 * 2);
  _Float16* WattT  = (_Float16*)alloc((size_t)P_ * P_ * 2);
  _Float16* Wlin2H = (_Float16*)alloc((size_t)P_ * 512 * 2);
  float*    Aproj  = (float*)   alloc((size_t)BT * P_ * 4);
  float*    Cproj  = (float*)   alloc((size_t)BT * P_ * 4);
  float*    Cmax   = (float*)   alloc((size_t)T_ * P_ * 4);
  float*    pooledF= (float*)   alloc((size_t)BT * P_ * 4);
  _Float16* pooledH= (_Float16*)alloc((size_t)BT * P_ * 2);
  _Float16* Gh     = (_Float16*)alloc((size_t)BT * P_ * 2);
  float*    scores = (float*)   alloc((size_t)B_ * T_ * T_ * 4);
  _Float16* attH   = (_Float16*)alloc((size_t)B_ * TPAD * SPAD * 2);
  _Float16* pooledT= (_Float16*)alloc((size_t)B_ * P_ * SPAD * 2);
  float*    attend = (float*)   alloc((size_t)BT * P_ * 4);
  _Float16* combH  = (_Float16*)alloc((size_t)BT * 2 * P_ * 2);
  _Float16* hidH   = (_Float16*)alloc((size_t)BT * P_ * 2);
  (void)alloc(4096);  // guard region for padded-tile overreads

  auto cvt = [&](const float* src, _Float16* dst, int rows, int sk, int dk) {
    const int n = rows * dk;
    f32_to_f16_pad<<<(n + 255) / 256, 256, 0, stream>>>(src, dst, rows, sk, dk);
  };
  auto gemm = [&](int NW, const _Float16* A, const _Float16* Bt, float* Cf, _Float16* Ch,
                  int M, int N, int K, int lda, int ldb, int ldc,
                  long sA, long sB, long sC, int nb, int Mv, int Nv,
                  const float* bias, int relu) {
    const long waves = (long)(M / 16) * (N / (16 * NW)) * nb;
    const int blocks = (int)((waves + 7) / 8);
    if (NW == 4)
      gemm_wmma<4><<<blocks, 256, 0, stream>>>(A, Bt, Cf, Ch, M, N, K, lda, ldb, ldc,
                                               sA, sB, sC, nb, Mv, Nv, bias, relu);
    else
      gemm_wmma<1><<<blocks, 256, 0, stream>>>(A, Bt, Cf, Ch, M, N, K, lda, ldb, ldc,
                                               sA, sB, sC, nb, Mv, Nv, bias, relu);
  };

  // --- weight prep (f32 -> f16, K-pad, W_att transpose) ---
  cvt(Wf0, Wih0F, G3, 258, K0P);  cvt(Wb0, Wih0B, G3, 258, K0P);
  cvt(Uf0, Whh0F, G3, U_, U_);    cvt(Ub0, Whh0B, G3, U_, U_);
  cvt(Wf1, Wih1F, G3, 512, 512);  cvt(Wb1, Wih1B, G3, 512, 512);
  cvt(Uf1, Whh1F, G3, U_, U_);    cvt(Ub1, Whh1B, G3, U_, U_);
  cvt(W_lin, WlinH, P_, 1024, 1024);
  cvt(W_lin2, Wlin2H, P_, 512, 512);
  watt_transpose<<<(P_ * P_ + 255) / 256, 256, 0, stream>>>(W_att, WattT);

  // --- embedding + concat -> X0 (BT x 288) f16 ---
  embed_concat<<<(BT * K0P + 255) / 256, 256, 0, stream>>>(words, is_pred, is_target, emb, X0);

  const size_t scanSmem = (size_t)BG * U_ * 2 + (size_t)BG * U_ * 4 + (size_t)BG * G3 * 4;

  // --- layer 0: xp = X0 @ Wih^T + bih, then scan -> H1 ---
  gemm(4, X0, Wih0F, xpFb, nullptr, BT, G3, K0P, K0P, K0P, G3, 0, 0, 0, 1, BT, G3, bif0, 0);
  gemm(4, X0, Wih0B, xpBb, nullptr, BT, G3, K0P, K0P, K0P, G3, 0, 0, 0, 1, BT, G3, bib0, 0);
  gru_scan<<<2 * NBG, 384, scanSmem, stream>>>(xpFb, xpBb, Whh0F, Whh0B, bhf0, bhb0, H1);

  // --- layer 1 ---
  gemm(4, H1, Wih1F, xpFb, nullptr, BT, G3, 512, 512, 512, G3, 0, 0, 0, 1, BT, G3, bif1, 0);
  gemm(4, H1, Wih1B, xpBb, nullptr, BT, G3, 512, 512, 512, G3, 0, 0, 0, 1, BT, G3, bib1, 0);
  gru_scan<<<2 * NBG, 384, scanSmem, stream>>>(xpFb, xpBb, Whh1F, Whh1B, bhf1, bhb1, H2);

  // --- A = H @ W1^T, C = H @ W2^T (W_lin split along K) ---
  gemm(4, H2, WlinH,       Aproj, nullptr, BT, P_, 512, 512, 1024, P_, 0, 0, 0, 1, BT, P_, nullptr, 0);
  gemm(4, H2, WlinH + 512, Cproj, nullptr, BT, P_, 512, 512, 1024, P_, 0, 0, 0, 1, BT, P_, nullptr, 0);

  // --- pooled = relu(A + max_b C + b_lin)  (batch-max pool, relu monotone) ---
  cmax_kernel<<<(T_ * P_ + 255) / 256, 256, 0, stream>>>(Cproj, Cmax);
  pooled_kernel<<<(BT * P_ + 255) / 256, 256, 0, stream>>>(Aproj, Cmax, b_lin, pooledF, pooledH);

  // --- attention: G = pooled @ W_att ; scores = G @ pooled^T (per batch) ---
  gemm(4, pooledH, WattT, nullptr, Gh, BT, P_, P_, P_, P_, P_, 0, 0, 0, 1, BT, P_, nullptr, 0);
  gemm(1, Gh, pooledH, scores, nullptr, TPAD, TPAD, P_, P_, P_, T_,
       (long)T_ * P_, (long)T_ * P_, (long)T_ * T_, B_, T_, T_, nullptr, 0);
  softmax_att<<<BT, 128, 0, stream>>>(scores, attH);

  // --- attend = att @ pooled (per batch; K padded to 128 with zeros) ---
  pooledT_kernel<<<(B_ * P_ * SPAD + 255) / 256, 256, 0, stream>>>(pooledF, pooledT);
  gemm(4, attH, pooledT, attend, nullptr, TPAD, P_, SPAD, SPAD, SPAD, P_,
       (long)TPAD * SPAD, (long)P_ * SPAD, (long)T_ * P_, B_, T_, P_, nullptr, 0);

  // --- comb = [pooled | attend]; hid = relu(comb @ W_lin2^T + b_lin2) ---
  comb_kernel<<<(BT * 2 * P_ + 255) / 256, 256, 0, stream>>>(pooledF, attend, combH);
  gemm(4, combH, Wlin2H, nullptr, hidH, BT, P_, 512, 512, 512, P_, 0, 0, 0, 1, BT, P_, b_lin2, 1);

  // --- logits + log_softmax ---
  final_logits<<<(BT + 127) / 128, 128, 0, stream>>>(hidH, W_out, b_out, (float*)d_out);
}
